// Up_62096637165752
// MI455X (gfx1250) — matrix operand order repaired and checked by reading, not compiled, not run on hardware
//
#include <hip/hip_runtime.h>

// MI455X (gfx1250): all conv GEMMs on v_wmma_f32_16x16x32_bf16, weight (B)
// fragments staged into LDS via global_load_async_to_lds_b128 (ASYNCcnt path),
// 8 waves per block sharing one staged weight tile per (tap, kblock).

typedef __attribute__((ext_vector_type(16))) __bf16 v16bf;
typedef __attribute__((ext_vector_type(8)))  float  v8f;
typedef __attribute__((ext_vector_type(4)))  unsigned int v4u;
typedef __attribute__((ext_vector_type(4)))  int v4i;

// async-to-LDS builtin operand types (per hipcc diagnostic: v4i in AS1 / AS3)
typedef __attribute__((address_space(1))) v4i* gv4i_p;
typedef __attribute__((address_space(3))) v4i* lv4i_p;

#define WB 2
#define WH 128
#define WW 128
#define CIN 256
#define CMID 128

#if defined(__has_builtin)
#if __has_builtin(__builtin_amdgcn_global_load_async_to_lds_b128) && \
    __has_builtin(__builtin_amdgcn_s_wait_asynccnt)
#define USE_ASYNC_LDS 1
#endif
#endif
#ifndef USE_ASYNC_LDS
#define USE_ASYNC_LDS 0
#endif

__device__ __forceinline__ void stage16(const void* g, void* l) {
#if USE_ASYNC_LDS
  __builtin_amdgcn_global_load_async_to_lds_b128((gv4i_p)g, (lv4i_p)l, 0, 0);
#else
  *(v4u*)l = *(const v4u*)g;
#endif
}
__device__ __forceinline__ void stage_wait() {
#if USE_ASYNC_LDS
  __builtin_amdgcn_s_wait_asynccnt(0);
#endif
}

__device__ __forceinline__ unsigned short f2bf(float f) {
  unsigned int u = __float_as_uint(f);
  u += 0x7FFFu + ((u >> 16) & 1u);   // round-to-nearest-even
  return (unsigned short)(u >> 16);
}
__device__ __forceinline__ float bf2f(unsigned short h) {
  return __uint_as_float(((unsigned int)h) << 16);
}

// ---------------------------------------------------------------------------
// Upsample x1 (per-reference coords) + concat with x2 -> NHWC bf16
// ---------------------------------------------------------------------------
__global__ void k_up_concat(const float* __restrict__ x1, const float* __restrict__ x2,
                            unsigned short* __restrict__ xbf) {
  long long idx = (long long)blockIdx.x * blockDim.x + threadIdx.x;
  const long long total = (long long)WB * WH * WW * CIN;
  if (idx >= total) return;
  int c = (int)(idx & (CIN - 1));
  long long p = idx >> 8;
  int xx = (int)(p % WW); p /= WW;
  int yy = (int)(p % WH); p /= WH;
  int b = (int)p;
  float v;
  if (c < 128) {
    v = x2[(((long long)b * 128 + c) * WH + yy) * WW + xx];
  } else {
    int cc = c - 128;
    const float sc = 63.0f / 127.0f;
    float cy = yy * sc, cx = xx * sc;
    int i0 = (int)floorf(cy); if (i0 < 0) i0 = 0; if (i0 > 62) i0 = 62;
    int j0 = (int)floorf(cx); if (j0 < 0) j0 = 0; if (j0 > 62) j0 = 62;
    float fy = cy - (float)i0, fx = cx - (float)j0;
    const float* base = x1 + ((long long)b * 128 + cc) * 64 * 64;
    float v00 = base[i0 * 64 + j0],       v01 = base[i0 * 64 + j0 + 1];
    float v10 = base[(i0 + 1) * 64 + j0], v11 = base[(i0 + 1) * 64 + j0 + 1];
    v = (v00 * (1.f - fy) + v10 * fy) * (1.f - fx) + (v01 * (1.f - fy) + v11 * fy) * fx;
  }
  xbf[idx] = f2bf(v);
}

// ---------------------------------------------------------------------------
// Pack OIHW fp32 weights into WMMA B-fragment order, bf16.
// Fragment = (tap, kb, nb): 512 bf16; lane L owns 16 contiguous bf16 (32B).
// B(32x16) element (k,n): lane = n + 16*(k>>4), slot j = k&15.
// ---------------------------------------------------------------------------
__global__ void k_pack(const float* __restrict__ w, unsigned short* __restrict__ wp,
                       int Cout, int Cin, int ksz, int KB, int NB) {
  int idx = blockIdx.x * blockDim.x + threadIdx.x;
  int total = ksz * ksz * KB * NB * 512;
  if (idx >= total) return;
  int j = idx & 15;
  int lane = (idx >> 4) & 31;
  int frag = idx >> 9;
  int nb = frag % NB; int t = frag / NB;
  int kb = t % KB; int tap = t / KB;
  int n = nb * 16 + (lane & 15);
  int k = kb * 32 + ((lane >> 4) << 4) + j;
  float v = 0.0f;
  if (n < Cout && k < Cin) v = w[((long long)n * Cin + k) * ksz * ksz + tap];
  wp[idx] = f2bf(v);
}

// ---------------------------------------------------------------------------
// Dense conv as tap-wise GEMM on WMMA. Block = 256 threads = 8 waves = one
// output row (8 x 16 pixels) x NB*16 channels. Per (tap,kb) the block stages
// the NB weight fragments into LDS with async-to-LDS copies; each wave then
// feeds its A fragment (two global_load_b128 from NHWC bf16) to WMMA.
// ---------------------------------------------------------------------------
template <int KB, int NB>
__global__ void __launch_bounds__(256)
k_conv_wmma(const unsigned short* __restrict__ xbf,
            const unsigned short* __restrict__ wp,
            const float* __restrict__ bias,
            float* __restrict__ out,
            int Cout, int ksz, int pad, int dil, int doClip) {
  const int Cin = KB * 32;
  __shared__ __align__(16) unsigned short sB[NB * 512];
  int tid  = threadIdx.x;
  int lane = tid & 31;
  int wave = tid >> 5;
  int m = lane & 15, laneHi = lane >> 4;
  int pix0 = (blockIdx.x * 8 + wave) * 16;
  int b = pix0 / (WH * WW);
  int rem = pix0 - b * WH * WW;
  int h = rem / WW;
  int w0 = rem - h * WW;
  int xcolBase = w0 + m;

  v8f acc[NB];
#pragma unroll
  for (int nb = 0; nb < NB; ++nb) acc[nb] = (v8f){0.f,0.f,0.f,0.f,0.f,0.f,0.f,0.f};

  const int taps = ksz * ksz;
  for (int tap = 0; tap < taps; ++tap) {
    int ky = tap / ksz, kx = tap - ky * ksz;
    int y  = h + ky * dil - pad;
    int xc = xcolBase + kx * dil - pad;
    bool inb = (y >= 0) && (y < WH) && (xc >= 0) && (xc < WW);
    int ys = inb ? y : 0, xs = inb ? xc : 0;
    const unsigned short* xrow = xbf + (long long)((b * WH + ys) * WW + xs) * Cin;
#pragma unroll
    for (int kb = 0; kb < KB; ++kb) {
      // --- stage NB weight fragments (NB*1024 bytes) into LDS ---
      __syncthreads();
      {
        const char* gsrc = (const char*)(wp + (((long long)(tap * KB + kb) * NB) << 9));
        char* ldst = (char*)sB;
        for (int off = tid * 16; off < NB * 1024; off += 256 * 16)
          stage16(gsrc + off, ldst + off);
        stage_wait();
      }
      __syncthreads();

      int k0 = kb * 32 + laneHi * 16;
      union { v16bf v; v4u u[2]; } A;
      if (inb) {
        A.u[0] = *(const v4u*)(xrow + k0);
        A.u[1] = *(const v4u*)(xrow + k0 + 8);
      } else {
        A.u[0] = (v4u){0u, 0u, 0u, 0u};
        A.u[1] = (v4u){0u, 0u, 0u, 0u};
      }
#pragma unroll
      for (int nb = 0; nb < NB; ++nb) {
        const unsigned short* bp = sB + (nb << 9) + lane * 16;
        union { v16bf v; v4u u[2]; } Bm;
        Bm.u[0] = *(const v4u*)(bp);
        Bm.u[1] = *(const v4u*)(bp + 8);
        acc[nb] = __builtin_amdgcn_wmma_f32_16x16x32_bf16(
            false, A.v, false, Bm.v, (short)0, acc[nb], false, false);
      }
    }
  }
  int n0 = lane & 15;
#pragma unroll
  for (int nb = 0; nb < NB; ++nb) {
    int n = nb * 16 + n0;
    if (n >= Cout) continue;
    float bv = bias[n];
#pragma unroll
    for (int r = 0; r < 8; ++r) {
      int mm = r + laneHi * 8;
      float v = acc[nb][r] + bv;
      if (doClip) v = fminf(fmaxf(v, -1.0f), 1.0f);
      out[(((long long)b * Cout + n) * WH + h) * WW + (w0 + mm)] = v;
    }
  }
}

// ---------------------------------------------------------------------------
// Deformable 3x3 conv (weights = packed conv0_w). Bilinear-gathered A fragments
// (masked corners per reference), blended in f32, rounded to bf16, WMMA.
// Same 8-wave / LDS-staged-B structure as k_conv_wmma.
// ---------------------------------------------------------------------------
__global__ void __launch_bounds__(256)
k_dconv_wmma(const unsigned short* __restrict__ xbf,
             const unsigned short* __restrict__ wp,
             const float* __restrict__ off,
             float* __restrict__ out, int dil) {
  __shared__ __align__(16) unsigned short sB[8 * 512];
  int tid  = threadIdx.x;
  int lane = tid & 31;
  int wave = tid >> 5;
  int m = lane & 15, laneHi = lane >> 4;
  int pix0 = (blockIdx.x * 8 + wave) * 16;
  int b = pix0 / (WH * WW);
  int rem = pix0 - b * WH * WW;
  int h = rem / WW;
  int w0 = rem - h * WW;
  int px = w0 + m;

  v8f acc[8];
#pragma unroll
  for (int nb = 0; nb < 8; ++nb) acc[nb] = (v8f){0.f,0.f,0.f,0.f,0.f,0.f,0.f,0.f};

  for (int tap = 0; tap < 9; ++tap) {
    int ky = tap / 3, kx = tap - ky * 3;
    long long obase = (((long long)b * 18 + 2 * tap) * WH + h) * WW + px;
    float oy = off[obase];
    float ox = off[obase + (long long)WH * WW];
    float ysf = (float)h  + (float)((ky - 1) * dil) + oy;
    float xsf = (float)px + (float)((kx - 1) * dil) + ox;
    float y0f = floorf(ysf), x0f = floorf(xsf);
    float wy = ysf - y0f, wx = xsf - x0f;
    int y0 = (int)y0f, x0 = (int)x0f;
    int y1i = y0 + 1, x1i = x0 + 1;
    bool vy0 = (y0 >= 0) && (y0 < WH), vy1 = (y1i >= 0) && (y1i < WH);
    bool vx0 = (x0 >= 0) && (x0 < WW), vx1 = (x1i >= 0) && (x1i < WW);
    float W00 = (vy0 && vx0) ? (1.f - wy) * (1.f - wx) : 0.f;
    float W01 = (vy0 && vx1) ? (1.f - wy) * wx : 0.f;
    float W10 = (vy1 && vx0) ? wy * (1.f - wx) : 0.f;
    float W11 = (vy1 && vx1) ? wy * wx : 0.f;
    int cy0 = min(max(y0, 0), WH - 1), cy1 = min(max(y1i, 0), WH - 1);
    int cx0 = min(max(x0, 0), WW - 1), cx1 = min(max(x1i, 0), WW - 1);
    const unsigned short* p00 = xbf + (long long)((b * WH + cy0) * WW + cx0) * CIN;
    const unsigned short* p01 = xbf + (long long)((b * WH + cy0) * WW + cx1) * CIN;
    const unsigned short* p10 = xbf + (long long)((b * WH + cy1) * WW + cx0) * CIN;
    const unsigned short* p11 = xbf + (long long)((b * WH + cy1) * WW + cx1) * CIN;

#pragma unroll
    for (int kb = 0; kb < 8; ++kb) {
      __syncthreads();
      {
        const char* gsrc = (const char*)(wp + (((long long)(tap * 8 + kb) * 8) << 9));
        char* ldst = (char*)sB;
        for (int off2 = tid * 16; off2 < 8 * 1024; off2 += 256 * 16)
          stage16(gsrc + off2, ldst + off2);
        stage_wait();
      }
      __syncthreads();

      int k0 = kb * 32 + laneHi * 16;
      union { v16bf v; v4u u[2]; unsigned short s[16]; } A, Q00, Q01, Q10, Q11;
      Q00.u[0] = *(const v4u*)(p00 + k0); Q00.u[1] = *(const v4u*)(p00 + k0 + 8);
      Q01.u[0] = *(const v4u*)(p01 + k0); Q01.u[1] = *(const v4u*)(p01 + k0 + 8);
      Q10.u[0] = *(const v4u*)(p10 + k0); Q10.u[1] = *(const v4u*)(p10 + k0 + 8);
      Q11.u[0] = *(const v4u*)(p11 + k0); Q11.u[1] = *(const v4u*)(p11 + k0 + 8);
#pragma unroll
      for (int e = 0; e < 16; ++e) {
        float s = W00 * bf2f(Q00.s[e]) + W01 * bf2f(Q01.s[e])
                + W10 * bf2f(Q10.s[e]) + W11 * bf2f(Q11.s[e]);
        A.s[e] = f2bf(s);
      }
#pragma unroll
      for (int nb = 0; nb < 8; ++nb) {
        const unsigned short* bp = sB + (nb << 9) + lane * 16;
        union { v16bf v; v4u u[2]; } Bm;
        Bm.u[0] = *(const v4u*)(bp);
        Bm.u[1] = *(const v4u*)(bp + 8);
        acc[nb] = __builtin_amdgcn_wmma_f32_16x16x32_bf16(
            false, A.v, false, Bm.v, (short)0, acc[nb], false, false);
      }
    }
  }
  int n0 = lane & 15;
#pragma unroll
  for (int nb = 0; nb < 8; ++nb) {
    int n = nb * 16 + n0;
#pragma unroll
    for (int r = 0; r < 8; ++r) {
      int mm = r + laneHi * 8;
      out[(((long long)b * CMID + n) * WH + h) * WW + (w0 + mm)] = acc[nb][r];
    }
  }
}

// ---------------------------------------------------------------------------
// Eval 1x1 conv + softmax + blend, plus deterministic per-block partial sums
// (blocks 0..63 = batch 0, 64..127 = batch 1).
// ---------------------------------------------------------------------------
__global__ void __launch_bounds__(256)
k_blend(const float* __restrict__ x0, const float* __restrict__ xd1,
        const float* __restrict__ xd2, const float* __restrict__ ew,
        const float* __restrict__ eb, float* __restrict__ y1,
        float* __restrict__ part) {
  int idx = blockIdx.x * 256 + threadIdx.x;   // pixel over B*H*W = 32768
  int b = idx >> 14;
  int p = idx & 16383;
  long long base = (long long)b * CMID * WH * WW + p;
  float s0 = eb[0], s1 = 0.f, s2 = 0.f;
  for (int c = 0; c < CMID; ++c) {
    float wv = ew[c];
    long long o = base + (long long)c * WH * WW;
    s0 += wv * x0[o];
    s1 += wv * xd1[o];
    s2 += wv * xd2[o];
  }
  float mx = fmaxf(s0, fmaxf(s1, s2));
  float e0 = expf(s0 - mx), e1 = expf(s1 - mx), e2 = expf(s2 - mx);
  float inv = 1.0f / (e0 + e1 + e2);
  e0 *= inv; e1 *= inv; e2 *= inv;
  float lsum = 0.f, lss = 0.f;
  for (int c = 0; c < CMID; ++c) {
    long long o = base + (long long)c * WH * WW;
    float v = e0 * x0[o] + e1 * xd1[o] + e2 * xd2[o];
    y1[o] = v;
    lsum += v; lss += v * v;
  }
  __shared__ float ss[256], sq[256];
  ss[threadIdx.x] = lsum; sq[threadIdx.x] = lss;
  __syncthreads();
  for (int st = 128; st > 0; st >>= 1) {
    if (threadIdx.x < st) { ss[threadIdx.x] += ss[threadIdx.x + st]; sq[threadIdx.x] += sq[threadIdx.x + st]; }
    __syncthreads();
  }
  if (threadIdx.x == 0) { part[blockIdx.x * 2] = ss[0]; part[blockIdx.x * 2 + 1] = sq[0]; }
}

// Deterministic partial sums over g (NCHW): block bl covers 16384 contiguous elems.
__global__ void __launch_bounds__(256)
k_partial(const float* __restrict__ g, float* __restrict__ part) {
  long long base = (long long)blockIdx.x * 16384;
  float s = 0.f, q = 0.f;
  for (int i = threadIdx.x; i < 16384; i += 256) {
    float v = g[base + i]; s += v; q += v * v;
  }
  __shared__ float ss[256], sq[256];
  ss[threadIdx.x] = s; sq[threadIdx.x] = q;
  __syncthreads();
  for (int st = 128; st > 0; st >>= 1) {
    if (threadIdx.x < st) { ss[threadIdx.x] += ss[threadIdx.x + st]; sq[threadIdx.x] += sq[threadIdx.x + st]; }
    __syncthreads();
  }
  if (threadIdx.x == 0) { part[blockIdx.x * 2] = ss[0]; part[blockIdx.x * 2 + 1] = sq[0]; }
}

__global__ void k_stats(const float* __restrict__ part, int nPerBatch, float count,
                        float* __restrict__ stats) {
  int b = threadIdx.x;
  if (b >= WB) return;
  float s = 0.f, q = 0.f;
  for (int i = 0; i < nPerBatch; ++i) {
    s += part[(b * nPerBatch + i) * 2];
    q += part[(b * nPerBatch + i) * 2 + 1];
  }
  float mu = s / count;
  float var = q / count - mu * mu;
  stats[b * 2] = mu;
  stats[b * 2 + 1] = rsqrtf(var + 1e-5f);
}

// LN + ReLU, NCHW f32 -> NHWC bf16 (input to final WMMA conv)
__global__ void k_ln_relu_tobf(const float* __restrict__ y1, const float* __restrict__ stats,
                               unsigned short* __restrict__ hbf) {
  long long idx = (long long)blockIdx.x * 256 + threadIdx.x;
  const long long total = (long long)WB * CMID * WH * WW;
  if (idx >= total) return;
  int b = (int)(idx / ((long long)CMID * WH * WW));
  long long r = idx % ((long long)CMID * WH * WW);
  int c = (int)(r / (WH * WW));
  int p = (int)(r % (WH * WW));
  int y = p / WW, x = p % WW;
  float v = (y1[idx] - stats[b * 2]) * stats[b * 2 + 1];
  v = fmaxf(v, 0.0f);
  hbf[((long long)(b * WH + y) * WW + x) * CMID + c] = f2bf(v);
}

// Final LN + ReLU -> output f32 NCHW
__global__ void k_ln_relu_out(const float* __restrict__ g, const float* __restrict__ stats,
                              float* __restrict__ outp) {
  long long idx = (long long)blockIdx.x * 256 + threadIdx.x;
  const long long total = (long long)WB * CMID * WH * WW;
  if (idx >= total) return;
  int b = (int)(idx / ((long long)CMID * WH * WW));
  float v = (g[idx] - stats[b * 2]) * stats[b * 2 + 1];
  outp[idx] = fmaxf(v, 0.0f);
}

// ---------------------------------------------------------------------------
extern "C" void kernel_launch(void* const* d_in, const int* in_sizes, int n_in,
                              void* d_out, int out_size, void* d_ws, size_t ws_size,
                              hipStream_t stream) {
  (void)in_sizes; (void)n_in; (void)out_size; (void)ws_size;
  const float* x1      = (const float*)d_in[0];
  const float* x2      = (const float*)d_in[1];
  const float* conv0_w = (const float*)d_in[2];
  const float* conv0_b = (const float*)d_in[3];
  const float* off1_w  = (const float*)d_in[4];
  const float* off1_b  = (const float*)d_in[5];
  const float* off2_w  = (const float*)d_in[6];
  const float* off2_b  = (const float*)d_in[7];
  const float* eval_w  = (const float*)d_in[8];
  const float* eval_b  = (const float*)d_in[9];
  const float* out_w   = (const float*)d_in[10];
  const float* out_b   = (const float*)d_in[11];
  float* outp = (float*)d_out;

  char* ws = (char*)d_ws;
  size_t cur = 0;
  auto alloc = [&](size_t bytes) -> char* {
    char* p = ws + cur;
    cur += (bytes + 255) & ~(size_t)255;
    return p;
  };
  unsigned short* xbf  = (unsigned short*)alloc((size_t)WB * WH * WW * CIN * 2);   // reused as y1
  float* x0            = (float*)alloc((size_t)WB * CMID * WH * WW * 4);           // reused as g
  float* xd1           = (float*)alloc((size_t)WB * CMID * WH * WW * 4);           // reused as hbf
  float* xd2           = (float*)alloc((size_t)WB * CMID * WH * WW * 4);
  float* obuf          = (float*)alloc((size_t)WB * 18 * WH * WW * 4);             // o1 then o2
  unsigned short* wp0  = (unsigned short*)alloc((size_t)9  * 8 * 8 * 512 * 2);
  unsigned short* wp1  = (unsigned short*)alloc((size_t)25 * 8 * 2 * 512 * 2);
  unsigned short* wp2  = (unsigned short*)alloc((size_t)49 * 8 * 2 * 512 * 2);
  unsigned short* wp3  = (unsigned short*)alloc((size_t)9  * 4 * 8 * 512 * 2);
  float* part          = (float*)alloc(256 * 2 * 4);
  float* stats         = (float*)alloc(4 * 4);
  float* y1            = (float*)xbf;            // xbf dead after deform convs
  unsigned short* hbf  = (unsigned short*)xd1;   // xd1 dead after blend
  float* g             = x0;                     // x0 dead after blend

  // 1) pack weights into WMMA B-fragment order
  { int t = 9  * 8 * 8 * 512; k_pack<<<(t + 255) / 256, 256, 0, stream>>>(conv0_w, wp0, 128, 256, 3, 8, 8); }
  { int t = 25 * 8 * 2 * 512; k_pack<<<(t + 255) / 256, 256, 0, stream>>>(off1_w,  wp1, 18,  256, 5, 8, 2); }
  { int t = 49 * 8 * 2 * 512; k_pack<<<(t + 255) / 256, 256, 0, stream>>>(off2_w,  wp2, 18,  256, 7, 8, 2); }
  { int t = 9  * 4 * 8 * 512; k_pack<<<(t + 255) / 256, 256, 0, stream>>>(out_w,   wp3, 128, 128, 3, 4, 8); }

  // 2) upsample + concat -> NHWC bf16
  { long long t = (long long)WB * WH * WW * CIN;
    k_up_concat<<<(unsigned)((t + 255) / 256), 256, 0, stream>>>(x1, x2, xbf); }

  const int MT8 = WB * WH * WW / (16 * 8);   // 256 blocks, 8 waves each

  // 3) conv0 (3x3, 256->128)
  k_conv_wmma<8, 8><<<MT8, 256, 0, stream>>>(xbf, wp0, conv0_b, x0, 128, 3, 1, 1, 0);
  // 4) off1 (5x5, 256->18, clip) ; deform conv dilate=1
  k_conv_wmma<8, 2><<<MT8, 256, 0, stream>>>(xbf, wp1, off1_b, obuf, 18, 5, 2, 1, 1);
  k_dconv_wmma<<<MT8, 256, 0, stream>>>(xbf, wp0, obuf, xd1, 1);
  // 5) off2 (7x7, 256->18, clip) ; deform conv dilate=2
  k_conv_wmma<8, 2><<<MT8, 256, 0, stream>>>(xbf, wp2, off2_b, obuf, 18, 7, 3, 1, 1);
  k_dconv_wmma<<<MT8, 256, 0, stream>>>(xbf, wp0, obuf, xd2, 2);

  // 6) eval + softmax + blend (+ LN1 partials)
  k_blend<<<128, 256, 0, stream>>>(x0, xd1, xd2, eval_w, eval_b, y1, part);
  k_stats<<<1, 32, 0, stream>>>(part, 64, 2097152.0f, stats);

  // 7) LN1 + ReLU -> bf16 NHWC
  { long long t = (long long)WB * CMID * WH * WW;
    k_ln_relu_tobf<<<(unsigned)((t + 255) / 256), 256, 0, stream>>>(y1, stats, hbf); }

  // 8) out conv (3x3, 128->128)
  k_conv_wmma<4, 8><<<MT8, 256, 0, stream>>>(hbf, wp3, out_b, g, 128, 3, 1, 1, 0);

  // 9) LN2 + ReLU -> output
  k_partial<<<256, 256, 0, stream>>>(g, part);
  k_stats<<<1, 32, 0, stream>>>(part, 128, 2097152.0f, stats);
  { long long t = (long long)WB * CMID * WH * WW;
    k_ln_relu_out<<<(unsigned)((t + 255) / 256), 256, 0, stream>>>(g, stats, outp); }
}
